// MultiHeadedAttention_54322746359918
// MI455X (gfx1250) — matrix operand order repaired and checked by reading, not compiled, not run on hardware
//
#include <hip/hip_runtime.h>

#define B_  2
#define S_  2048
#define D_  1024
#define H_  16
#define DH_ 64
#define M_  (B_ * S_)   // 4096
#define K_  D_          // 1024
#define N_  D_          // 1024

typedef __attribute__((ext_vector_type(16))) __bf16    v16bf;
typedef __attribute__((ext_vector_type(8)))  float     v8f;
typedef __attribute__((ext_vector_type(4)))  int       v4i;
typedef __attribute__((ext_vector_type(4)))  unsigned  u32x4;
typedef __attribute__((ext_vector_type(8)))  int       i32x8;
typedef __attribute__((ext_vector_type(4)))  int       i32x4;

union FragB16 { v16bf v; unsigned u[8]; };

static __device__ inline unsigned short f2bf(float x) {
  union { float f; unsigned u; } c; c.f = x;
  unsigned r = c.u + 0x7FFFu + ((c.u >> 16) & 1u);   // round-to-nearest-even
  return (unsigned short)(r >> 16);
}
static __device__ inline unsigned pack2bf(float a, float b) {
  return (unsigned)f2bf(a) | ((unsigned)f2bf(b) << 16);
}

// ---------------- CDNA5 async global->LDS path (probe-guarded) ----------------
#if __has_builtin(__builtin_amdgcn_global_load_async_to_lds_b128)
#define USE_ASYNC_LDS 1
typedef v4i __attribute__((address_space(1))) * g_v4i_p;   // global (device) AS
typedef v4i __attribute__((address_space(3))) * l_v4i_p;   // LDS AS
static __device__ inline void async_b128(const unsigned short* g, unsigned short* l) {
  __builtin_amdgcn_global_load_async_to_lds_b128((g_v4i_p)(void*)g, (l_v4i_p)(void*)l, 0, 0);
}
static __device__ inline void wait_async0() {
#if __has_builtin(__builtin_amdgcn_s_wait_asynccnt)
  __builtin_amdgcn_s_wait_asynccnt(0);
#else
  asm volatile("s_wait_asynccnt 0" ::: "memory");
#endif
}
#else
#define USE_ASYNC_LDS 0
static __device__ inline void wait_async0() {}
#endif

// Copies one thread's 32B slice (two b128) of a bf16 tile (fallback staging).
static __device__ inline void stage2(const unsigned short* g, unsigned short* l) {
#if USE_ASYNC_LDS
  async_b128(g,     l);
  async_b128(g + 8, l + 8);
#else
  ((uint4*)l)[0] = ((const uint4*)g)[0];
  ((uint4*)l)[1] = ((const uint4*)g)[1];
#endif
}

// ---------------- CDNA5 Tensor Data Mover path (probe-guarded) ----------------
#if __has_builtin(__builtin_amdgcn_tensor_load_to_lds)
#define USE_TDM 1
#if defined(__has_include)
#if __has_include(<hip/amd_detail/amd_gfx1250_TDM.h>)
#define TDM_6ARG 1
#else
#define TDM_6ARG 0
#endif
#else
#define TDM_6ARG 0
#endif

typedef void __attribute__((address_space(3))) * lds_void_p;
static __device__ inline unsigned lds_addr_of(void* p) {
  return (unsigned)(size_t)(lds_void_p)p;   // LDS byte offset
}

// D# group 0 (ISA 8.3): count=1 | lds_addr[63:32] | global_addr[120:64] | type=2
static __device__ inline u32x4 tdm_group0(unsigned lds_byte, const void* gaddr) {
  const unsigned long long ga = (unsigned long long)(size_t)gaddr;
  u32x4 g;
  g.x = 1u;                                                  // count=1
  g.y = lds_byte;                                            // lds_addr
  g.z = (unsigned)ga;                                        // global_addr[31:0]
  g.w = ((unsigned)(ga >> 32) & 0x01FFFFFFu) | (2u << 30);   // [56:32] | type=2
  return g;
}
// D# group 1 (ISA 8.4) for a 2D tile, data_size=2B, no padding, no multicast.
// tensor dims == tile dims (no OOB), stride in elements.
static __device__ inline i32x8 tdm_group1_2d(unsigned td0, unsigned td1,
                                             unsigned t0, unsigned t1,
                                             unsigned long long stride0) {
  i32x8 g;
  g[0] = (int)(1u << 16);                                        // data_size=1 (2B)
  g[1] = (int)((td0 & 0xFFFFu) << 16);                           // tensor_dim0[15:0]
  g[2] = (int)(((td0 >> 16) & 0xFFFFu) | ((td1 & 0xFFFFu) << 16));
  g[3] = (int)(((td1 >> 16) & 0xFFFFu) | ((t0 & 0xFFFFu) << 16)); // tile_dim0
  g[4] = (int)(t1 & 0xFFFFu);                                    // tile_dim1; dim2=0
  g[5] = (int)(unsigned)(stride0 & 0xFFFFFFFFu);                 // dim0_stride[31:0]
  g[6] = (int)(unsigned)((stride0 >> 32) & 0xFFFFu);             // dim0_stride[47:32]
  g[7] = 0;
  return g;
}
static __device__ inline void tdm_load(u32x4 g0, i32x8 g1) {
  const i32x4 z4 = {};
#if TDM_6ARG
  const i32x8 z8 = {};
  __builtin_amdgcn_tensor_load_to_lds(g0, g1, z4, z4, z8, 0);
#else
  __builtin_amdgcn_tensor_load_to_lds(g0, g1, z4, z4, 0);
#endif
}
static __device__ inline void wait_stage0() {
#if __has_builtin(__builtin_amdgcn_s_wait_tensorcnt)
  __builtin_amdgcn_s_wait_tensorcnt((short)0);
#else
  asm volatile("s_wait_tensorcnt 0" ::: "memory");
#endif
}
#else
#define USE_TDM 0
static __device__ inline void wait_stage0() { wait_async0(); }
#endif

// ---- Fragment loaders per CDNA5 ISA 7.12.2 VGPR layouts (wave32) ----

// A fragment, 16x32 bf16, element A[m][k] = base[m*ldm + k].
// lanes 0-15: m=lane, K={0..7,16..23}; lanes 16-31: m=lane-16, K={8..15,24..31}.
static __device__ inline v16bf load_frag_a(const unsigned short* base, int ldm, int lane) {
  const int m    = lane & 15;
  const int koff = (lane < 16) ? 0 : 8;
  FragB16 f;
  const unsigned short* p = base + m * ldm + koff;
#pragma unroll
  for (int j = 0; j < 8; ++j) {
    const int k = (j < 4) ? (2 * j) : (16 + 2 * (j - 4));
    f.u[j] = *(const unsigned*)(p + k);  // k even -> dword-aligned pair
  }
  return f.v;
}

// B fragment, 32x16 bf16, K contiguous in memory: B[k][n] = base[n*ldn + k].
// lanes 0-15: n=lane, K={0..7,16..23}; lanes 16-31: n=lane-16, K={8..15,24..31}.
static __device__ inline v16bf load_frag_b_kc(const unsigned short* base, int ldn, int lane) {
  const int n    = lane & 15;
  const int koff = (lane < 16) ? 0 : 8;
  FragB16 f;
  const unsigned short* p = base + n * ldn + koff;
#pragma unroll
  for (int j = 0; j < 8; ++j) f.u[j] = *(const unsigned*)(p + ((j < 4) ? (2 * j) : (16 + 2 * (j - 4))));
  return f.v;
}

static __device__ inline v8f wmma_bf16(v16bf a, v16bf b, v8f c) {
  return __builtin_amdgcn_wmma_f32_16x16x32_bf16(false, a, false, b, (short)0, c, false, false);
}

// ---------------- Kernel 0: fp32 -> bf16 convert (one pass) ----------------
__global__ void __launch_bounds__(256) cvt_bf16_kernel(const float* __restrict__ in,
                                                       unsigned short* __restrict__ out,
                                                       int n8) {
  const int i = blockIdx.x * 256 + threadIdx.x;   // 8 elements per thread
  if (i < n8) {
    const float4 f0 = ((const float4*)in)[2 * i];
    const float4 f1 = ((const float4*)in)[2 * i + 1];
    uint4 p;
    p.x = pack2bf(f0.x, f0.y); p.y = pack2bf(f0.z, f0.w);
    p.z = pack2bf(f1.x, f1.y); p.w = pack2bf(f1.z, f1.w);
    ((uint4*)out)[i] = p;
  }
}

// ---------------- Shared bf16 GEMM main loop (double-buffered, TDM/async) ----
// C(128x128) += A(128xK) * W(128-row x K)^T, K-step 32.
// 8 waves: wave (wid&3) owns 32 M-rows, (wid>>2) owns 64 N-cols.
static __device__ inline void gemm_bf16_tile(const unsigned short* __restrict__ A,
                                             const unsigned short* __restrict__ Wb,
                                             unsigned short As[2][128 * 32],
                                             unsigned short Bs[2][128 * 32],
                                             v8f acc[2][4], int mtile, int ntile)
{
  const int t    = threadIdx.x;
  const int lane = t & 31;
  const int wid  = t >> 5;
  const int wm   = (wid & 3) * 32;
  const int wn   = (wid >> 2) * 64;
  const int row  = t >> 1;          // 0..127 staging row (fallback path)
  const int cb   = (t & 1) << 4;    // 0 or 16 staging col base
  const int NT   = K_ / 32;

  const unsigned short* agB = A  + (size_t)(mtile + row) * K_ + cb;
  const unsigned short* bgB = Wb + (size_t)(ntile + row) * K_ + cb;
  (void)agB; (void)bgB;

  // issue tile kt into buffer buf
  auto issue = [&](int kt, int buf) {
#if USE_TDM
    // one 128x32 tile per descriptor; TDM DMA engine fills LDS while waves WMMA
    if (wid == 0)
      tdm_load(tdm_group0(lds_addr_of(&As[buf][0]), A + (size_t)mtile * K_ + kt * 32),
               tdm_group1_2d(32, 128, 32, 128, K_));
    else if (wid == 1)
      tdm_load(tdm_group0(lds_addr_of(&Bs[buf][0]), Wb + (size_t)ntile * K_ + kt * 32),
               tdm_group1_2d(32, 128, 32, 128, K_));
#else
    stage2(agB + kt * 32, &As[buf][row * 32 + cb]);
    stage2(bgB + kt * 32, &Bs[buf][row * 32 + cb]);
#endif
  };

  issue(0, 0);
  for (int kt = 0; kt < NT; ++kt) {
    const int buf = kt & 1;
    wait_stage0();          // issuing waves: tile kt transfers complete
    __syncthreads();        // everyone's tile kt is in LDS; prev compute done
    if (kt + 1 < NT) issue(kt + 1, buf ^ 1);   // overlap next loads with WMMA

    const v16bf a0 = load_frag_a(&As[buf][(wm +  0) * 32], 32, lane);
    const v16bf a1 = load_frag_a(&As[buf][(wm + 16) * 32], 32, lane);
#pragma unroll
    for (int ni = 0; ni < 4; ++ni) {
      const v16bf b = load_frag_b_kc(&Bs[buf][(wn + ni * 16) * 32], 32, lane);
      acc[0][ni] = wmma_bf16(a0, b, acc[0][ni]);
      acc[1][ni] = wmma_bf16(a1, b, acc[1][ni]);
    }
  }
}

// ---------------- Kernel 1: fused QKV projection ----------------
// y = x @ W.T + b (all bf16 in).  Q,K stored [B,H,S,DH]; V stored transposed
// [B,H,DH,S] so the PV B-fragment has contiguous K.
__global__ void __launch_bounds__(256) qkv_gemm_kernel(
    const unsigned short* __restrict__ xb,
    const unsigned short* __restrict__ wqb, const float* __restrict__ bq,
    const unsigned short* __restrict__ wkb, const float* __restrict__ bk,
    const unsigned short* __restrict__ wvb, const float* __restrict__ bv,
    unsigned short* __restrict__ qo, unsigned short* __restrict__ ko,
    unsigned short* __restrict__ vo)
{
  __shared__ unsigned short As[2][128 * 32];
  __shared__ unsigned short Bs[2][128 * 32];

  const unsigned short* W;
  const float* bias;
  unsigned short* out;
  if (blockIdx.z == 0)      { W = wqb; bias = bq; out = qo; }
  else if (blockIdx.z == 1) { W = wkb; bias = bk; out = ko; }
  else                      { W = wvb; bias = bv; out = vo; }
  const bool vmode = (blockIdx.z == 2);

  const int ntile = blockIdx.x * 128;
  const int mtile = blockIdx.y * 128;

  v8f acc[2][4];
  const v8f zero = {};
#pragma unroll
  for (int i = 0; i < 2; ++i)
#pragma unroll
    for (int j = 0; j < 4; ++j) acc[i][j] = zero;

  gemm_bf16_tile(xb, W, As, Bs, acc, mtile, ntile);

  const int lane = threadIdx.x & 31;
  const int wid  = threadIdx.x >> 5;
  const int wm = (wid & 3) * 32, wn = (wid >> 2) * 64;
  const int ln = lane & 15;
  const int mb = (lane < 16) ? 0 : 8;
#pragma unroll
  for (int mi = 0; mi < 2; ++mi)
#pragma unroll
    for (int ni = 0; ni < 4; ++ni) {
      const int gn = ntile + wn + ni * 16 + ln;
      const float bb = bias[gn];
      const int h  = gn >> 6;   // / DH_
      const int dh = gn & 63;
#pragma unroll
      for (int r = 0; r < 8; ++r) {
        const int gm = mtile + wm + mi * 16 + mb + r;
        const int bi = gm >> 11;      // / S_
        const int s  = gm & 2047;
        const size_t idx = vmode
            ? ((size_t)(bi * H_ + h) * DH_ + dh) * S_ + s     // V^T layout
            : ((size_t)(bi * H_ + h) * S_ + s) * DH_ + dh;    // Q/K layout
        out[idx] = f2bf(acc[mi][ni][r] + bb);
      }
    }
}

// ---------------- Kernel 2: flash attention ----------------
// Per (b,h): softmax(Q K^T / sqrt(DH)) V with online softmax.
// Block = 128 query rows (8 waves x 16 rows); 64-key blocks double-buffered.
__global__ void __launch_bounds__(256) attn_kernel(
    const unsigned short* __restrict__ q, const unsigned short* __restrict__ k,
    const unsigned short* __restrict__ v, unsigned short* __restrict__ o)
{
  __shared__ unsigned short Ks[2][64 * 64];
  __shared__ unsigned short Vs[2][64 * 64];   // V^T tile: [dh][key]
  __shared__ unsigned short Ps[128 * 72];     // padded pitch, wave-private rows

  const int t = threadIdx.x, lane = t & 31, wid = t >> 5;
  const int qt = blockIdx.x;       // 0..15 query tiles
  const int bh = blockIdx.y;       // 0..31 (b*H + h)
  const size_t head_off = (size_t)bh * S_ * DH_;
  const unsigned short* qh = q + head_off;
  const unsigned short* kh = k + head_off;
  const unsigned short* vh = v + head_off;    // V^T: [DH][S]

  const int m0 = qt * 128 + wid * 16;
  const v16bf qa0 = load_frag_a(qh + (size_t)m0 * DH_ +  0, DH_, lane);
  const v16bf qa1 = load_frag_a(qh + (size_t)m0 * DH_ + 32, DH_, lane);

  v8f oacc[4];
  const v8f zero = {};
#pragma unroll
  for (int df = 0; df < 4; ++df) oacc[df] = zero;
  float mi_r[8], li_r[8];
#pragma unroll
  for (int r = 0; r < 8; ++r) { mi_r[r] = -3.0e38f; li_r[r] = 0.f; }

  const int ln = lane & 15;
  const int mb = (lane < 16) ? 0 : 8;
  const int srow = t >> 2;          // 0..63 staging row (fallback)
  const int scb  = (t & 3) << 4;    // staging col base
  const float CE = 0.125f * 1.44269504088896340736f;  // (1/sqrt(DH)) * log2(e)

  const unsigned short* kgB = kh + (size_t)srow * DH_ + scb;
  const unsigned short* vgB = vh + (size_t)srow * S_;
  (void)kgB; (void)vgB;

  auto issue = [&](int kb, int buf) {
#if USE_TDM
    if (wid == 0)       // K block: 64 rows x 64 cols, row stride DH
      tdm_load(tdm_group0(lds_addr_of(&Ks[buf][0]), kh + (size_t)kb * 64 * DH_),
               tdm_group1_2d(64, 64, 64, 64, DH_));
    else if (wid == 1)  // V^T block: 64 dh-rows x 64 key-cols, row stride S
      tdm_load(tdm_group0(lds_addr_of(&Vs[buf][0]), vh + kb * 64),
               tdm_group1_2d(64, 64, 64, 64, S_));
#else
    stage2(kgB + (size_t)kb * 64 * DH_, &Ks[buf][srow * 64 + scb]);
    stage2(vgB + kb * 64 + scb,         &Vs[buf][srow * 64 + scb]);
#endif
  };

  const int NB = S_ / 64;
  issue(0, 0);
  for (int kb = 0; kb < NB; ++kb) {
    const int buf = kb & 1;
    wait_stage0();
    __syncthreads();
    if (kb + 1 < NB) issue(kb + 1, buf ^ 1);   // prefetch next K/V block

    // scores: 16x64 per wave, S[m][n] = sum_d Q[m][d] * K[n][d]
    v8f sc[4];
#pragma unroll
    for (int nf = 0; nf < 4; ++nf) sc[nf] = zero;
#pragma unroll
    for (int nf = 0; nf < 4; ++nf) {
      const v16bf b0 = load_frag_b_kc(&Ks[buf][nf * 16 * 64 +  0], 64, lane);
      sc[nf] = wmma_bf16(qa0, b0, sc[nf]);
      const v16bf b1 = load_frag_b_kc(&Ks[buf][nf * 16 * 64 + 32], 64, lane);
      sc[nf] = wmma_bf16(qa1, b1, sc[nf]);
    }

    // online softmax: rows live in (VGPR index r, lane half); reduce across 16 lanes
#pragma unroll
    for (int r = 0; r < 8; ++r) {
      float mx = fmaxf(fmaxf(sc[0][r], sc[1][r]), fmaxf(sc[2][r], sc[3][r]));
      mx = fmaxf(mx, __shfl_xor(mx, 1, 32));
      mx = fmaxf(mx, __shfl_xor(mx, 2, 32));
      mx = fmaxf(mx, __shfl_xor(mx, 4, 32));
      mx = fmaxf(mx, __shfl_xor(mx, 8, 32));
      const float mnew = fmaxf(mi_r[r], mx);
      const float al   = exp2f((mi_r[r] - mnew) * CE);
      mi_r[r] = mnew;
      float rs = 0.f;
#pragma unroll
      for (int nf = 0; nf < 4; ++nf) {
        const float p = exp2f((sc[nf][r] - mnew) * CE);
        sc[nf][r] = p;
        rs += p;
      }
      rs += __shfl_xor(rs, 1, 32);
      rs += __shfl_xor(rs, 2, 32);
      rs += __shfl_xor(rs, 4, 32);
      rs += __shfl_xor(rs, 8, 32);
      li_r[r] = li_r[r] * al + rs;
#pragma unroll
      for (int df = 0; df < 4; ++df) oacc[df][r] *= al;
      // spill P row to LDS (C-layout -> A-layout re-fragmentation).
      // Wave-private rows: per-wave in-order DS ops make a barrier unnecessary.
#pragma unroll
      for (int nf = 0; nf < 4; ++nf)
        Ps[(wid * 16 + mb + r) * 72 + nf * 16 + ln] = f2bf(sc[nf][r]);
    }

    // O += P(16x64) * V(64x64); V^T tile gives K-contiguous B fragments
#pragma unroll
    for (int kk = 0; kk < 2; ++kk) {
      const v16bf pa = load_frag_a(&Ps[wid * 16 * 72 + kk * 32], 72, lane);
#pragma unroll
      for (int df = 0; df < 4; ++df) {
        const v16bf vb = load_frag_b_kc(&Vs[buf][(df * 16) * 64 + kk * 32], 64, lane);
        oacc[df] = wmma_bf16(pa, vb, oacc[df]);
      }
    }
  }

  // epilogue: normalize and write combined-head [B,S,D] bf16
  const int bi = bh / H_, h = bh % H_;
#pragma unroll
  for (int df = 0; df < 4; ++df) {
    const int col = h * DH_ + df * 16 + ln;
#pragma unroll
    for (int r = 0; r < 8; ++r) {
      const int s = qt * 128 + wid * 16 + mb + r;
      const float val = oacc[df][r] * (1.0f / li_r[r]);
      o[(size_t)(bi * S_ + s) * D_ + col] = f2bf(val);
    }
  }
}

// ---------------- Kernel 3: output projection ----------------
// out(fp32) = O(bf16) @ Wo.T + bo
__global__ void __launch_bounds__(256) out_proj_kernel(
    const unsigned short* __restrict__ Oin, const unsigned short* __restrict__ wob,
    const float* __restrict__ bo, float* __restrict__ out)
{
  __shared__ unsigned short As[2][128 * 32];
  __shared__ unsigned short Bs[2][128 * 32];

  const int ntile = blockIdx.x * 128;
  const int mtile = blockIdx.y * 128;

  v8f acc[2][4];
  const v8f zero = {};
#pragma unroll
  for (int i = 0; i < 2; ++i)
#pragma unroll
    for (int j = 0; j < 4; ++j) acc[i][j] = zero;

  gemm_bf16_tile(Oin, wob, As, Bs, acc, mtile, ntile);

  const int lane = threadIdx.x & 31;
  const int wid  = threadIdx.x >> 5;
  const int wm = (wid & 3) * 32, wn = (wid >> 2) * 64;
  const int ln = lane & 15;
  const int mb = (lane < 16) ? 0 : 8;
#pragma unroll
  for (int mi = 0; mi < 2; ++mi)
#pragma unroll
    for (int ni = 0; ni < 4; ++ni) {
      const int gn = ntile + wn + ni * 16 + ln;
      const float bb = bo[gn];
#pragma unroll
      for (int r = 0; r < 8; ++r) {
        const int gm = mtile + wm + mi * 16 + mb + r;
        out[(size_t)gm * N_ + gn] = acc[mi][ni][r] + bb;
      }
    }
}

// ---------------- Host launcher ----------------
extern "C" void kernel_launch(void* const* d_in, const int* in_sizes, int n_in,
                              void* d_out, int out_size, void* d_ws, size_t ws_size,
                              hipStream_t stream) {
  (void)in_sizes; (void)n_in; (void)out_size; (void)ws_size;
  const float* x  = (const float*)d_in[0];
  const float* Wq = (const float*)d_in[1]; const float* bq = (const float*)d_in[2];
  const float* Wk = (const float*)d_in[3]; const float* bk = (const float*)d_in[4];
  const float* Wv = (const float*)d_in[5]; const float* bv = (const float*)d_in[6];
  const float* Wo = (const float*)d_in[7]; const float* bo = (const float*)d_in[8];
  float* out = (float*)d_out;

  // workspace layout (bf16 elements)
  const size_t nX = (size_t)M_ * D_;   // 4M
  const size_t nW = (size_t)D_ * D_;   // 1M
  unsigned short* xb  = (unsigned short*)d_ws;
  unsigned short* wqb = xb  + nX;
  unsigned short* wkb = wqb + nW;
  unsigned short* wvb = wkb + nW;
  unsigned short* wob = wvb + nW;
  unsigned short* qws = wob + nW;
  unsigned short* kws = qws + nX;
  unsigned short* vws = kws + nX;      // stored transposed [B,H,DH,S]
  unsigned short* ows = vws + nX;      // total ~48 MB

  // 0) one-pass fp32 -> bf16 conversion (memory-bound, ~2us at 23.3 TB/s)
  cvt_bf16_kernel<<<dim3((unsigned)(nX / 8 / 256)), 256, 0, stream>>>(x,  xb,  (int)(nX / 8));
  cvt_bf16_kernel<<<dim3((unsigned)(nW / 8 / 256)), 256, 0, stream>>>(Wq, wqb, (int)(nW / 8));
  cvt_bf16_kernel<<<dim3((unsigned)(nW / 8 / 256)), 256, 0, stream>>>(Wk, wkb, (int)(nW / 8));
  cvt_bf16_kernel<<<dim3((unsigned)(nW / 8 / 256)), 256, 0, stream>>>(Wv, wvb, (int)(nW / 8));
  cvt_bf16_kernel<<<dim3((unsigned)(nW / 8 / 256)), 256, 0, stream>>>(Wo, wob, (int)(nW / 8));

  // 1) fused QKV projection (z selects Q/K/V)
  dim3 g1(N_ / 128, M_ / 128, 3);
  qkv_gemm_kernel<<<g1, 256, 0, stream>>>(xb, wqb, bq, wkb, bk, wvb, bv, qws, kws, vws);

  // 2) flash attention
  dim3 g2(S_ / 128, B_ * H_);
  attn_kernel<<<g2, 256, 0, stream>>>(qws, kws, vws, ows);

  // 3) output projection -> fp32 d_out
  dim3 g3(N_ / 128, M_ / 128);
  out_proj_kernel<<<g3, 256, 0, stream>>>(ows, wob, bo, out);
}
